// TemporalGCN_74045236183579
// MI455X (gfx1250) — matrix operand order repaired and checked by reading, not compiled, not run on hardware
//
#include <hip/hip_runtime.h>
#include <hip/hip_bf16.h>
#include <math.h>

typedef _Float16 f16;
typedef __attribute__((ext_vector_type(16))) _Float16 v16h;
typedef __attribute__((ext_vector_type(8)))  _Float16 v8h;
typedef __attribute__((ext_vector_type(8)))  float    v8f;

#define B_  8
#define W_  12
#define N_  200
#define H_  64
#define E_  5
#define F_  6
#define G4H 256
#define BN  1600
#define LN_EPS 1e-5f
#define LOG2E 1.44269504088896f

// ---------- fast device math (raw v_exp_f32 / v_rcp_f32; inputs are bounded) ----------
static __device__ inline float fast_sigm(float x) {
  return __builtin_amdgcn_rcpf(1.0f + __builtin_amdgcn_exp2f(-LOG2E * x));
}
static __device__ inline float fast_tanh(float x) {
  return 1.0f - 2.0f * __builtin_amdgcn_rcpf(1.0f + __builtin_amdgcn_exp2f(2.0f * LOG2E * x));
}

static __device__ inline v8f splat8(float x) {
  v8f v;
#pragma unroll
  for (int i = 0; i < 8; ++i) v[i] = x;
  return v;
}
static __device__ inline v16h zero16h() {
  v16h v;
#pragma unroll
  for (int i = 0; i < 16; ++i) v[i] = (f16)0.0f;
  return v;
}
static __device__ inline v16h combine8(v8h lo, v8h hi) {
  return __builtin_shufflevector(lo, hi, 0,1,2,3,4,5,6,7,8,9,10,11,12,13,14,15);
}

// WMMA wrapper: D = A(16x32 f16) * B(32x16 f16) + C(f32)
static __device__ inline v8f wmma16(v16h a, v16h b, v8f c) {
  return __builtin_amdgcn_wmma_f32_16x16x32_f16(false, a, false, b, (short)0, c, false, false);
}

// Load A-fragment (16x32, K-tile kt) from a row-major [16][64] f16 buffer.
// A layout (ISA 7.12.2): lane m(0..15): halves = K{0..7,16..23}; lane m+16: K{8..15,24..31}
static __device__ inline v16h ldA_rowmajor(const f16* base, int lane, int kt) {
  const int m = lane & 15;
  const int g = (lane >> 4) & 1;
  const f16* row = base + m * 64 + kt * 32 + g * 8;
  v8h lo = *(const v8h*)(row);        // K = kt*32 + 8g + (0..7)
  v8h hi = *(const v8h*)(row + 16);   // K = kt*32 + 16 + 8g + (0..7)
  return combine8(lo, hi);
}

// Load A-fragment from globally pre-packed hA: per row 64 halves laid out [kt][g][16]
static __device__ inline v16h ldA_packed(const f16* hA, int row, int lane, int kt) {
  const int g = (lane >> 4) & 1;
  return *(const v16h*)(hA + (size_t)row * 64 + kt * 32 + g * 16);
}

// Load B-fragment (32x16): packed tiles of [32 lanes][16 halves]
static __device__ inline v16h ldB(const f16* Bp, int tile, int lane) {
  return *(const v16h*)(Bp + (size_t)tile * 512 + lane * 16);
}

// ---------- kernel 1: pack a weight matrix W[nrows][ncols] into f16 B-tiles ----------
// B[k][n] = W[n][k]; tileIdx = nt*ktiles + kt; lane<16 -> k in kt*32 + {0..7,16..23}
__global__ void packB_kernel(const float* __restrict__ W, int nrows, int ncols,
                             int ktiles, int ntiles, f16* __restrict__ dst) {
  int tid = blockIdx.x * 256 + threadIdx.x;
  int total = ktiles * ntiles * 512;
  if (tid >= total) return;
  int e = tid & 15;
  int lane = (tid >> 4) & 31;
  int tile = tid >> 9;
  int kt = tile % ktiles;
  int nt = tile / ktiles;
  int k = kt * 32 + ((lane >> 4) & 1) * 8 + (e & 7) + (e >> 3) * 16;
  int n = nt * 16 + (lane & 15);
  float v = (k < ncols && n < nrows) ? W[(size_t)n * ncols + k] : 0.0f;
  dst[tid] = (f16)v;
}

// ---------- kernel 2: x @ Wih0^T + (bih0+bhh0), packed in WMMA C-fragment order ----------
// layout: [wt(100)][t(12)][nt(16)][lane(32)][vr(8)] f32 ; m = vr + 8*(lane>>4), n = (lane&15)+16*nt
__global__ void xproj_kernel(const float* __restrict__ x,
                             const float* __restrict__ Wih0,
                             const float* __restrict__ bih0,
                             const float* __restrict__ bhh0,
                             float* __restrict__ xp) {
  long tid = (long)blockIdx.x * 256 + threadIdx.x;
  if (tid >= (long)100 * 12 * 16 * 32 * 8) return;
  long idx = tid;
  int vr = (int)(idx & 7);  idx >>= 3;
  int lane = (int)(idx & 31); idx >>= 5;
  int nt = (int)(idx & 15); idx >>= 4;
  int t = (int)(idx % 12);  idx /= 12;
  int wt = (int)idx;
  int seq = wt * 16 + vr + 8 * (lane >> 4);
  int n = (lane & 15) + 16 * nt;
  int b = seq / N_, nn = seq % N_;
  const float* xr = x + (((long)b * W_ + t) * N_ + nn) * F_;
  const float* wr = Wih0 + n * F_;
  float acc = bih0[n] + bhh0[n];
#pragma unroll
  for (int f = 0; f < F_; ++f) acc += xr[f] * wr[f];
  xp[tid] = acc;
}

// ---------- kernel 3: two-layer LSTM, WMMA recurrent GEMMs ----------
__global__ __launch_bounds__(128) void lstm_kernel(
    const float* __restrict__ xp,
    const f16* __restrict__ pBhh0, const f16* __restrict__ pBih1, const f16* __restrict__ pBhh1,
    const float* __restrict__ bih1, const float* __restrict__ bhh1,
    float* __restrict__ hout) {
  __shared__ __align__(32) f16 hb0[4][16][64];
  __shared__ __align__(32) f16 hb1[4][16][64];
  const int w = threadIdx.x >> 5;
  const int lane = threadIdx.x & 31;
  const int wt = blockIdx.x * 4 + w;   // 0..99, exact

  for (int idx = lane; idx < 16 * 64; idx += 32) {
    hb0[w][idx >> 6][idx & 63] = (f16)0.0f;
    hb1[w][idx >> 6][idx & 63] = (f16)0.0f;
  }
  __syncthreads();

  v8f c0[4], c1[4];
#pragma unroll
  for (int i = 0; i < 4; ++i) { c0[i] = splat8(0.0f); c1[i] = splat8(0.0f); }

  for (int t = 0; t < W_; ++t) {
    // Opaque zero, re-materialized each iteration: makes the weight-fragment
    // addresses loop-variant so LICM cannot hoist ~100KB of B-fragments into
    // registers (which caused scratch spill/reload in the inner loop).
    int tofs = 0;
    asm volatile("" : "+v"(tofs));
    const f16* B0  = pBhh0 + tofs;
    const f16* Bi1 = pBih1 + tofs;
    const f16* Bh1 = pBhh1 + tofs;
    const float* bi1 = bih1 + tofs;
    const float* bh1 = bhh1 + tofs;

    // ---- layer 0: g = xproj + h0 @ Whh0^T ----
    v16h a0 = ldA_rowmajor(&hb0[w][0][0], lane, 0);
    v16h a1 = ldA_rowmajor(&hb0[w][0][0], lane, 1);
#pragma unroll
    for (int ht = 0; ht < 4; ++ht) {
      v8f gf[4];
#pragma unroll
      for (int gs = 0; gs < 4; ++gs) {
        int nt = gs * 4 + ht;
        long base = ((((long)wt * 12 + t) * 16 + nt) * 32 + lane) * 8;
        v8f c = *(const v8f*)(xp + base);
        c = wmma16(a0, ldB(B0, nt * 2 + 0, lane), c);
        c = wmma16(a1, ldB(B0, nt * 2 + 1, lane), c);
        gf[gs] = c;
      }
#pragma unroll
      for (int r = 0; r < 8; ++r) {
        float ig = fast_sigm(gf[0][r]);
        float fg = fast_sigm(gf[1][r]);
        float gg = fast_tanh(gf[2][r]);
        float og = fast_sigm(gf[3][r]);
        float cn = fg * c0[ht][r] + ig * gg;
        c0[ht][r] = cn;
        float hv = og * fast_tanh(cn);
        int m = r + 8 * (lane >> 4);
        int n = 16 * ht + (lane & 15);
        hb0[w][m][n] = (f16)hv;
      }
    }
    __syncthreads();
    // ---- layer 1: g = bias1 + y0 @ Wih1^T + h1 @ Whh1^T ----
    v16h y0a = ldA_rowmajor(&hb0[w][0][0], lane, 0);
    v16h y0b = ldA_rowmajor(&hb0[w][0][0], lane, 1);
    v16h h1a = ldA_rowmajor(&hb1[w][0][0], lane, 0);
    v16h h1b = ldA_rowmajor(&hb1[w][0][0], lane, 1);
#pragma unroll
    for (int ht = 0; ht < 4; ++ht) {
      v8f gf[4];
#pragma unroll
      for (int gs = 0; gs < 4; ++gs) {
        int nt = gs * 4 + ht;
        int n = (lane & 15) + 16 * nt;
        v8f c = splat8(bi1[n] + bh1[n]);
        c = wmma16(y0a, ldB(Bi1, nt * 2 + 0, lane), c);
        c = wmma16(y0b, ldB(Bi1, nt * 2 + 1, lane), c);
        c = wmma16(h1a, ldB(Bh1, nt * 2 + 0, lane), c);
        c = wmma16(h1b, ldB(Bh1, nt * 2 + 1, lane), c);
        gf[gs] = c;
      }
#pragma unroll
      for (int r = 0; r < 8; ++r) {
        float ig = fast_sigm(gf[0][r]);
        float fg = fast_sigm(gf[1][r]);
        float gg = fast_tanh(gf[2][r]);
        float og = fast_sigm(gf[3][r]);
        float cn = fg * c1[ht][r] + ig * gg;
        c1[ht][r] = cn;
        float hv = og * fast_tanh(cn);
        int m = r + 8 * (lane >> 4);
        int n = 16 * ht + (lane & 15);
        hb1[w][m][n] = (f16)hv;
      }
    }
    __syncthreads();
  }
  // write final h1 (f32)
  for (int idx = lane; idx < 16 * 64; idx += 32) {
    int m = idx >> 6, n = idx & 63;
    hout[(size_t)(wt * 16 + m) * 64 + n] = (float)hb1[w][m][n];
  }
}

// ---------- kernel 4: s_e[r][e] = sum_j adj*edge, s_a[r] = sum_j adj ----------
__global__ __launch_bounds__(64) void esum_kernel(const float* __restrict__ adj,
                                                  const float* __restrict__ edge,
                                                  float* __restrict__ s_e,
                                                  float* __restrict__ s_a) {
  __shared__ float red[6][64];
  int r = blockIdx.x;
  int tid = threadIdx.x;
  int b = r / N_, i = r % N_;
  const float* arow = adj + ((size_t)b * N_ + i) * N_;
  const float* erow = edge + ((((long)b * W_ + (W_ - 1)) * N_ + i) * N_) * E_;
  float acc[6] = {0, 0, 0, 0, 0, 0};
  for (int j = tid; j < N_; j += 64) {
    float a = arow[j];
    const float* ep = erow + (size_t)j * E_;
#pragma unroll
    for (int e = 0; e < E_; ++e) acc[e] += a * ep[e];
    acc[5] += a;
  }
#pragma unroll
  for (int c = 0; c < 6; ++c) red[c][tid] = acc[c];
  __syncthreads();
  for (int s = 32; s > 0; s >>= 1) {
    if (tid < s) {
#pragma unroll
      for (int c = 0; c < 6; ++c) red[c][tid] += red[c][tid + s];
    }
    __syncthreads();
  }
  if (tid < E_) s_e[(size_t)r * E_ + tid] = red[tid][0];
  if (tid == 5) s_a[r] = red[5][0];
}

// ---------- kernel 5: one GCN iteration + layernorm + relu (optionally pack hA) ----------
__global__ __launch_bounds__(64) void gcn_kernel(const float* __restrict__ h_in,
                                                 float* __restrict__ h_out,
                                                 const float* __restrict__ s_e,
                                                 const float* __restrict__ s_a,
                                                 const float* __restrict__ gcnW,
                                                 const float* __restrict__ gcnb,
                                                 const float* __restrict__ epW,
                                                 const float* __restrict__ epb,
                                                 const float* __restrict__ lng,
                                                 const float* __restrict__ lnb,
                                                 int iter, f16* __restrict__ hA) {
  __shared__ float hrow[64];
  __shared__ float red[64];
  int r = blockIdx.x;
  int n = threadIdx.x;
  hrow[n] = h_in[(size_t)r * H_ + n];
  __syncthreads();
  const float* Wr = gcnW + ((size_t)iter * H_ + n) * H_;
  float acc = gcnb[iter * H_ + n];
#pragma unroll 8
  for (int k = 0; k < H_; ++k) acc += hrow[k] * Wr[k];
  const float* ew = epW + ((size_t)iter * H_ + n) * E_;
  float msg = s_a[r] * epb[iter * H_ + n];
#pragma unroll
  for (int e = 0; e < E_; ++e) msg += s_e[(size_t)r * E_ + e] * ew[e];
  float u = acc + msg;
  // layernorm over 64
  red[n] = u;
  __syncthreads();
  for (int s = 32; s > 0; s >>= 1) { if (n < s) red[n] += red[n + s]; __syncthreads(); }
  float mu = red[0] * (1.0f / 64.0f);
  __syncthreads();
  float d = u - mu;
  red[n] = d * d;
  __syncthreads();
  for (int s = 32; s > 0; s >>= 1) { if (n < s) red[n] += red[n + s]; __syncthreads(); }
  float var = red[0] * (1.0f / 64.0f);
  float y = d * rsqrtf(var + LN_EPS) * lng[iter * H_ + n] + lnb[iter * H_ + n];
  float h = y > 0.0f ? y : 0.0f;
  h_out[(size_t)r * H_ + n] = h;
  if (hA) {  // pack into WMMA A-fragment order: per row [kt][g][16]
    int kt = n >> 5, kl = n & 31, g, e;
    if (kl < 8)       { g = 0; e = kl; }
    else if (kl < 16) { g = 1; e = kl - 8; }
    else if (kl < 24) { g = 0; e = kl - 8; }
    else              { g = 1; e = kl - 16; }
    hA[(size_t)r * 64 + kt * 32 + g * 16 + e] = (f16)h;
  }
}

// ---------- kernel 6: pairwise edge MLP (WMMA), sigmoid output ----------
__global__ __launch_bounds__(128) void mlp_kernel(const f16* __restrict__ hA,
                                                  const float* __restrict__ edge,
                                                  const f16* __restrict__ pW1,
                                                  const float* __restrict__ b1,
                                                  const f16* __restrict__ pW2,
                                                  const float* __restrict__ b2,
                                                  const float* __restrict__ w3,
                                                  const float* __restrict__ b3,
                                                  float* __restrict__ out) {
  __shared__ __align__(32) f16 zb[4][16][64];
  __shared__ float z2b[4][16][32];
  const int w = threadIdx.x >> 5;
  const int lane = threadIdx.x & 31;
  const int tile = blockIdx.x * 4 + w;   // < 20800 exactly
  const int TJ = 13;                     // ceil(200/16)
  int pair = tile / TJ;
  int jt = tile % TJ;
  int b = pair / N_, i = pair % N_;
  int j0 = jt * 16;
  int ri = b * N_ + i;

  // A fragments, K = [h_i(64) | h_j(64) | edge(5) pad to 32]
  v16h A[5];
  A[0] = ldA_packed(hA, ri, lane, 0);
  A[1] = ldA_packed(hA, ri, lane, 1);
  int jm = j0 + (lane & 15);
  int jcl = jm < N_ ? jm : (N_ - 1);
  int rj = b * N_ + jcl;
  A[2] = ldA_packed(hA, rj, lane, 0);
  A[3] = ldA_packed(hA, rj, lane, 1);
  v16h ae = zero16h();
  if (lane < 16) {  // lanes<16 hold K-tile-local k = 0..7 in halves 0..7; only k<5 nonzero
    const float* ep = edge + ((((long)b * W_ + (W_ - 1)) * N_ + i) * N_ + jcl) * E_;
    ae[0] = (f16)ep[0]; ae[1] = (f16)ep[1]; ae[2] = (f16)ep[2];
    ae[3] = (f16)ep[3]; ae[4] = (f16)ep[4];
  }
  A[4] = ae;

  const int ncol = lane & 15;
  // layer 1: 133(->160) -> 64, relu
#pragma unroll
  for (int nt = 0; nt < 4; ++nt) {
    v8f c = splat8(b1[ncol + 16 * nt]);
#pragma unroll
    for (int kt = 0; kt < 5; ++kt) c = wmma16(A[kt], ldB(pW1, nt * 5 + kt, lane), c);
#pragma unroll
    for (int r = 0; r < 8; ++r) {
      float v = c[r] > 0.0f ? c[r] : 0.0f;
      zb[w][r + 8 * (lane >> 4)][ncol + 16 * nt] = (f16)v;
    }
  }
  __syncthreads();
  // layer 2: 64 -> 32, relu
  v16h za0 = ldA_rowmajor(&zb[w][0][0], lane, 0);
  v16h za1 = ldA_rowmajor(&zb[w][0][0], lane, 1);
#pragma unroll
  for (int nt = 0; nt < 2; ++nt) {
    v8f c = splat8(b2[ncol + 16 * nt]);
    c = wmma16(za0, ldB(pW2, nt * 2 + 0, lane), c);
    c = wmma16(za1, ldB(pW2, nt * 2 + 1, lane), c);
#pragma unroll
    for (int r = 0; r < 8; ++r) {
      float v = c[r] > 0.0f ? c[r] : 0.0f;
      z2b[w][r + 8 * (lane >> 4)][ncol + 16 * nt] = v;
    }
  }
  __syncthreads();
  // layer 3: 32 -> 1, sigmoid
  if (lane < 16) {
    int j = j0 + lane;
    if (j < N_) {
      float acc = b3[0];
#pragma unroll
      for (int n = 0; n < 32; ++n) acc += z2b[w][lane][n] * w3[n];
      out[((size_t)(b * N_ + i)) * N_ + j] = fast_sigm(acc);
    }
  }
}

// ---------- host launch ----------
extern "C" void kernel_launch(void* const* d_in, const int* in_sizes, int n_in,
                              void* d_out, int out_size, void* d_ws, size_t ws_size,
                              hipStream_t stream) {
  (void)in_sizes; (void)n_in; (void)out_size; (void)ws_size;
  const float* node = (const float*)d_in[0];
  const float* edge = (const float*)d_in[1];
  const float* adj  = (const float*)d_in[2];
  const float* Wih0 = (const float*)d_in[3];
  const float* Whh0 = (const float*)d_in[4];
  const float* bih0 = (const float*)d_in[5];
  const float* bhh0 = (const float*)d_in[6];
  const float* Wih1 = (const float*)d_in[7];
  const float* Whh1 = (const float*)d_in[8];
  const float* bih1 = (const float*)d_in[9];
  const float* bhh1 = (const float*)d_in[10];
  const float* gcnW = (const float*)d_in[11];
  const float* gcnb = (const float*)d_in[12];
  const float* epW  = (const float*)d_in[13];
  const float* epb  = (const float*)d_in[14];
  const float* lng  = (const float*)d_in[15];
  const float* lnb  = (const float*)d_in[16];
  const float* W1   = (const float*)d_in[17];
  const float* b1   = (const float*)d_in[18];
  const float* W2   = (const float*)d_in[19];
  const float* b2   = (const float*)d_in[20];
  const float* W3   = (const float*)d_in[21];
  const float* b3   = (const float*)d_in[22];
  float* out = (float*)d_out;

  char* ws = (char*)d_ws;
  size_t off = 0;
  auto take = [&](size_t bytes) -> void* {
    void* p = ws + off;
    off = (off + bytes + 255) & ~(size_t)255;
    return p;
  };
  float* xp    = (float*)take((size_t)100 * 12 * 16 * 32 * 8 * 4); // 19.66 MB
  f16* pBhh0   = (f16*)take((size_t)32 * 512 * 2);
  f16* pBih1   = (f16*)take((size_t)32 * 512 * 2);
  f16* pBhh1   = (f16*)take((size_t)32 * 512 * 2);
  f16* pW1     = (f16*)take((size_t)20 * 512 * 2);
  f16* pW2     = (f16*)take((size_t)4 * 512 * 2);
  f16* hA      = (f16*)take((size_t)BN * 64 * 2);
  float* h_a   = (float*)take((size_t)BN * 64 * 4);
  float* h_b   = (float*)take((size_t)BN * 64 * 4);
  float* s_e   = (float*)take((size_t)BN * E_ * 4);
  float* s_a   = (float*)take((size_t)BN * 4);

  // pack weights into WMMA B-fragment layouts (f16)
  packB_kernel<<<(32 * 512 + 255) / 256, 256, 0, stream>>>(Whh0, 256, 64, 2, 16, pBhh0);
  packB_kernel<<<(32 * 512 + 255) / 256, 256, 0, stream>>>(Wih1, 256, 64, 2, 16, pBih1);
  packB_kernel<<<(32 * 512 + 255) / 256, 256, 0, stream>>>(Whh1, 256, 64, 2, 16, pBhh1);
  packB_kernel<<<(20 * 512 + 255) / 256, 256, 0, stream>>>(W1, 64, 133, 5, 4, pW1);
  packB_kernel<<<(4 * 512 + 255) / 256, 256, 0, stream>>>(W2, 32, 64, 2, 2, pW2);

  // input projection for LSTM layer 0 (tiny K=6), pre-packed into C-fragment order
  xproj_kernel<<<19200, 256, 0, stream>>>(node, Wih0, bih0, bhh0, xp);

  // two-layer LSTM (WMMA)
  lstm_kernel<<<25, 128, 0, stream>>>(xp, pBhh0, pBih1, pBhh1, bih1, bhh1, h_a);

  // weighted edge-sum (shared by both GCN iterations)
  esum_kernel<<<BN, 64, 0, stream>>>(adj, edge, s_e, s_a);

  // GCN iterations with layernorm+relu; second pass packs hA for the MLP
  gcn_kernel<<<BN, 64, 0, stream>>>(h_a, h_b, s_e, s_a, gcnW, gcnb, epW, epb, lng, lnb, 0, (f16*)nullptr);
  gcn_kernel<<<BN, 64, 0, stream>>>(h_b, h_a, s_e, s_a, gcnW, gcnb, epW, epb, lng, lnb, 1, hA);

  // pairwise edge MLP (WMMA) -> sigmoid logits
  mlp_kernel<<<5200, 128, 0, stream>>>(hA, edge, pW1, b1, pW2, b2, W3, b3, out);
}
